// Net_1881195676352
// MI455X (gfx1250) — matrix operand order repaired and checked by reading, not compiled, not run on hardware
//
#include <hip/hip_runtime.h>

typedef __attribute__((ext_vector_type(16))) __bf16 v16bf;
typedef __attribute__((ext_vector_type(8)))  float  v8f;

#define B_ROWS   32768
#define K_DIM    1200
#define H_DIM    256
#define KC       32
#define NUM_KC   38          // ceil(1200/32); tail zero-padded via guards
#define NSTEPS   25
#define BETA     0.9f
#define THR      1.0f
#define LDS_STRIDE 40        // padded col stride (bf16 elems): conflict-free b128 reads

#define WAVES_PER_BLOCK 8
#define ROWS_PER_WAVE   16
#define ROWS_PER_BLOCK  (WAVES_PER_BLOCK * ROWS_PER_WAVE)   // 128

union BFrag { uint4 q[2]; v16bf v; };

__launch_bounds__(256)
__global__ void snn_fused_kernel(const float* __restrict__ x,
                                 const float* __restrict__ W1,
                                 const float* __restrict__ b1,
                                 const float* __restrict__ W2,
                                 const float* __restrict__ b2,
                                 float* __restrict__ out)
{
    // Double-buffered W1 K-chunk, bf16 hi/lo planes, layout [col][k], stride 40
    __shared__ __align__(16) __bf16 ldsHi[2][H_DIM * LDS_STRIDE];   // 2 x 20 KB
    __shared__ __align__(16) __bf16 ldsLo[2][H_DIM * LDS_STRIDE];   // 2 x 20 KB

    const int tid  = threadIdx.x;
    const int wave = tid >> 5;
    const int lane = tid & 31;
    const int m    = lane & 15;   // row (A) / col-in-tile (B,C)
    const int g    = lane >> 4;   // half-wave group
    const int rowBase = blockIdx.x * ROWS_PER_BLOCK + wave * ROWS_PER_WAVE;

    const float* __restrict__ xrow = x + (size_t)(rowBase + m) * K_DIM;

    // ---- pipelined loaders (one K-chunk ahead, registers) -----------------
    auto loadW = [&](int kc, float4 (&w)[8]) {
        const int kbase = kc * KC;
        #pragma unroll
        for (int it = 0; it < 8; ++it) {
            const int idx = tid + it * 256;      // 0..2047
            const int col = idx >> 3;            // 0..255
            const int k4  = (idx & 7) * 4;       // 0..28
            if (kbase + k4 + 4 <= K_DIM)         // 1200 % 4 == 0: run-granular
                w[it] = *(const float4*)(W1 + (size_t)col * K_DIM + kbase + k4);
            else
                w[it] = make_float4(0.f, 0.f, 0.f, 0.f);
        }
    };
    auto storeW = [&](int buf, const float4 (&w)[8]) {
        #pragma unroll
        for (int it = 0; it < 8; ++it) {
            const int idx = tid + it * 256;
            const int col = idx >> 3;
            const int k4  = (idx & 7) * 4;
            const float vv[4] = {w[it].x, w[it].y, w[it].z, w[it].w};
            unsigned int hp[2], lp[2];
            #pragma unroll
            for (int p = 0; p < 2; ++p) {
                const float f0 = vv[2 * p], f1 = vv[2 * p + 1];
                const __bf16 h0 = (__bf16)f0, h1 = (__bf16)f1;
                const __bf16 l0 = (__bf16)(f0 - (float)h0);
                const __bf16 l1 = (__bf16)(f1 - (float)h1);
                hp[p] = (unsigned)__builtin_bit_cast(unsigned short, h0) |
                        ((unsigned)__builtin_bit_cast(unsigned short, h1) << 16);
                lp[p] = (unsigned)__builtin_bit_cast(unsigned short, l0) |
                        ((unsigned)__builtin_bit_cast(unsigned short, l1) << 16);
            }
            *(uint2*)&ldsHi[buf][col * LDS_STRIDE + k4] = make_uint2(hp[0], hp[1]);
            *(uint2*)&ldsLo[buf][col * LDS_STRIDE + k4] = make_uint2(lp[0], lp[1]);
        }
    };
    auto loadA = [&](int kc, float (&a)[16]) {
        const int kbase = kc * KC;
        const int k0 = kbase + g * 8;         // 1200 % 8 == 0: run-granular
        const int k1 = kbase + 16 + g * 8;
        if (k0 + 8 <= K_DIM) {
            float4 u0 = *(const float4*)(xrow + k0);
            float4 u1 = *(const float4*)(xrow + k0 + 4);
            a[0]=u0.x; a[1]=u0.y; a[2]=u0.z; a[3]=u0.w;
            a[4]=u1.x; a[5]=u1.y; a[6]=u1.z; a[7]=u1.w;
        } else {
            #pragma unroll
            for (int i = 0; i < 8; ++i) a[i] = 0.f;
        }
        if (k1 + 8 <= K_DIM) {
            float4 u0 = *(const float4*)(xrow + k1);
            float4 u1 = *(const float4*)(xrow + k1 + 4);
            a[8]=u0.x;  a[9]=u0.y;  a[10]=u0.z; a[11]=u0.w;
            a[12]=u1.x; a[13]=u1.y; a[14]=u1.z; a[15]=u1.w;
        } else {
            #pragma unroll
            for (int i = 8; i < 16; ++i) a[i] = 0.f;
        }
    };

    v8f acc[16];
    #pragma unroll
    for (int t = 0; t < 16; ++t)
        #pragma unroll
        for (int i = 0; i < 8; ++i) acc[t][i] = 0.0f;

    // ---------------- GEMM: cur1 = x @ W1.T  (bf16 hi/lo split, f32 accum) --
    float4 wreg[8];
    float  areg[16];
    loadW(0, wreg);
    loadA(0, areg);

    for (int kc = 0; kc < NUM_KC; ++kc) {
        const int buf = kc & 1;

        storeW(buf, wreg);        // stage chunk kc -> LDS (hi/lo bf16)
        loadW(kc + 1, wreg);      // prefetch chunk kc+1 (guards: no-op past end)

        v16bf ah, al;             // A fragment for chunk kc
        #pragma unroll
        for (int i = 0; i < 16; ++i) {
            const __bf16 h = (__bf16)areg[i];
            ah[i] = h;
            al[i] = (__bf16)(areg[i] - (float)h);
        }
        loadA(kc + 1, areg);      // prefetch next A fragment
        __builtin_prefetch(xrow + (kc + 2) * KC + g * 8, 0, 0);

        __syncthreads();          // single barrier: store(kc) -> compute(kc);
                                  // double buffer protects compute(kc) vs store(kc+2)

        #pragma unroll
        for (int t = 0; t < 16; ++t) {
            const int col = t * 16 + m;
            const __bf16* ph = &ldsHi[buf][col * LDS_STRIDE + g * 8];
            const __bf16* pl = &ldsLo[buf][col * LDS_STRIDE + g * 8];
            BFrag bh, bl;
            bh.q[0] = *(const uint4*)(ph);
            bh.q[1] = *(const uint4*)(ph + 16);
            bl.q[0] = *(const uint4*)(pl);
            bl.q[1] = *(const uint4*)(pl + 16);
            acc[t] = __builtin_amdgcn_wmma_f32_16x16x32_bf16(
                         false, ah, false, bh.v, (short)0, acc[t], false, false);
            acc[t] = __builtin_amdgcn_wmma_f32_16x16x32_bf16(
                         false, al, false, bh.v, (short)0, acc[t], false, false);
            acc[t] = __builtin_amdgcn_wmma_f32_16x16x32_bf16(
                         false, ah, false, bl.v, (short)0, acc[t], false, false);
        }
    }

    // ---------------- bias b1 ---------------------------------------------
    #pragma unroll
    for (int t = 0; t < 16; ++t) {
        const float bias = b1[t * 16 + m];
        #pragma unroll
        for (int r = 0; r < 8; ++r) acc[t][r] += bias;
    }

    // ---------------- temporal SNN loop (registers only) -------------------
    float w2a[16], w2b[16];
    #pragma unroll
    for (int t = 0; t < 16; ++t) {
        w2a[t] = W2[t * 16 + m];            // W2[0][col]
        w2b[t] = W2[H_DIM + t * 16 + m];    // W2[1][col]
    }
    const float b20 = b2[0], b21 = b2[1];

    float mem1[16][8];
    #pragma unroll
    for (int t = 0; t < 16; ++t)
        #pragma unroll
        for (int r = 0; r < 8; ++r) mem1[t][r] = 0.f;

    float mem2a[8], mem2b[8], acc2a[8], acc2b[8];
    #pragma unroll
    for (int r = 0; r < 8; ++r) { mem2a[r]=0.f; mem2b[r]=0.f; acc2a[r]=0.f; acc2b[r]=0.f; }

    for (int step = 0; step < NSTEPS; ++step) {
        float p0[8], p1[8];
        #pragma unroll
        for (int r = 0; r < 8; ++r) { p0[r] = 0.f; p1[r] = 0.f; }

        // mem1 update + spike + partial GEMV against W2 (this lane's 16 cols)
        #pragma unroll
        for (int t = 0; t < 16; ++t) {
            #pragma unroll
            for (int r = 0; r < 8; ++r) {
                float mv = mem1[t][r];
                const float rst = (mv > THR) ? THR : 0.f;
                mv = fmaf(BETA, mv, acc[t][r]) - rst;
                mem1[t][r] = mv;
                const float spk = (mv > THR) ? 1.f : 0.f;
                p0[r] = fmaf(spk, w2a[t], p0[r]);
                p1[r] = fmaf(spk, w2b[t], p1[r]);
            }
        }

        // reduce across the 16 column-lanes of each half-wave, then layer 2
        #pragma unroll
        for (int r = 0; r < 8; ++r) {
            float s0 = p0[r], s1 = p1[r];
            s0 += __shfl_xor(s0, 8, 32);  s1 += __shfl_xor(s1, 8, 32);
            s0 += __shfl_xor(s0, 4, 32);  s1 += __shfl_xor(s1, 4, 32);
            s0 += __shfl_xor(s0, 2, 32);  s1 += __shfl_xor(s1, 2, 32);
            s0 += __shfl_xor(s0, 1, 32);  s1 += __shfl_xor(s1, 1, 32);
            const float c0 = s0 + b20;
            const float c1 = s1 + b21;
            const float r0 = (mem2a[r] > THR) ? THR : 0.f;
            const float r1 = (mem2b[r] > THR) ? THR : 0.f;
            mem2a[r] = fmaf(BETA, mem2a[r], c0) - r0;
            mem2b[r] = fmaf(BETA, mem2b[r], c1) - r1;
            acc2a[r] += (mem2a[r] > THR) ? 1.f : 0.f;
            acc2b[r] += (mem2b[r] > THR) ? 1.f : 0.f;
        }
    }

    // ---------------- write out [B,2] --------------------------------------
    if (m == 0) {
        #pragma unroll
        for (int r = 0; r < 8; ++r) {
            const int row = rowBase + r + 8 * g;
            out[row * 2 + 0] = acc2a[r];
            out[row * 2 + 1] = acc2b[r];
        }
    }
}

extern "C" void kernel_launch(void* const* d_in, const int* in_sizes, int n_in,
                              void* d_out, int out_size, void* d_ws, size_t ws_size,
                              hipStream_t stream) {
    (void)in_sizes; (void)n_in; (void)out_size; (void)d_ws; (void)ws_size;
    const float* x  = (const float*)d_in[0];
    const float* W1 = (const float*)d_in[1];
    const float* b1 = (const float*)d_in[2];
    const float* W2 = (const float*)d_in[3];
    const float* b2 = (const float*)d_in[4];
    float* out = (float*)d_out;

    dim3 grid(B_ROWS / ROWS_PER_BLOCK);   // 256 blocks
    dim3 block(256);                      // 8 wave32s
    snn_fused_kernel<<<grid, block, 0, stream>>>(x, W1, b1, W2, b2, out);
}
